// HiGNN_48979807043681
// MI455X (gfx1250) — compile-verified, matrix operand order checked by reading, not run on hardware
//
#include <hip/hip_runtime.h>
#include <hip/hip_bf16.h>
#include <math.h>

#define N_NODES  100000
#define N_EDGES  500000
#define N_GRAPHS 4096
#define IN_CH    46
#define EDGE_DIM 10
#define HID      128
#define SLICES   4
#define FA_HID   32
#define EDGE_TILES (N_EDGES / 16)

typedef __bf16 bf16_t;
typedef __attribute__((ext_vector_type(16))) __bf16 v16bf;
typedef __attribute__((ext_vector_type(8)))  float  v8f;

union Frag16 { v16bf v; bf16_t h[16]; uint4 q[2]; };

// CDNA5 16-bit A/B fragment layout (ISA 7.12.2):
// lanes 0-15: elements 0..7 -> K 0..7, elements 8..15 -> K 16..23
// lanes 16-31: elements 0..7 -> K 8..15, elements 8..15 -> K 24..31
__device__ __forceinline__ int kmap_e(int lane, int e) {
  int hi = (lane >> 4) & 1;
  int base = (e < 8) ? (hi ? 8 : 0) : (hi ? 24 : 16);
  return base + (e & 7);
}

// A fragment: row fixed per lane (m = lane&15); two 16B global loads.
__device__ __forceinline__ void load_a_frag(Frag16 &a, const bf16_t* row, int lane, int kbase) {
  int off = kbase + ((lane & 16) ? 8 : 0);
  a.q[0] = *(const uint4*)(row + off);
  a.q[1] = *(const uint4*)(row + off + 16);
}

__device__ __forceinline__ void load_b_frag(Frag16 &b, const bf16_t* sBase, int fragIdx, int lane) {
  const uint4* p = (const uint4*)(sBase + fragIdx * 512 + lane * 16);
  b.q[0] = p[0]; b.q[1] = p[1];
}

__device__ __forceinline__ v8f wmma_bf16(const Frag16 &a, const Frag16 &b, v8f c) {
  return __builtin_amdgcn_wmma_f32_16x16x32_bf16(false, a.v, false, b.v, (short)0, c, false, false);
}

__device__ __forceinline__ void atomicMaxFloat(float* addr, float val) {
  unsigned* ua = (unsigned*)addr;
  unsigned old = *ua;
  while (true) {
    float f = __uint_as_float(old);
    if (f >= val) break;
    unsigned assumed = old;
    old = atomicCAS(ua, assumed, __float_as_uint(val));
    if (old == assumed) break;
  }
}

// ---------------------------------------------------------------------------
// C[M,128] = A[M,128] @ B[128,128]; A bf16 row-major, B staged from f32 to
// LDS pre-swizzled into WMMA fragment order. B-fragments double-buffered so
// ds_load latency overlaps WMMA issue. C written bf16.
// ---------------------------------------------------------------------------
__global__ void __launch_bounds__(256, 1)
gemm128_kernel(const bf16_t* __restrict__ A, const float* __restrict__ Bf,
               bf16_t* __restrict__ C, int M)
{
  __shared__ __align__(16) bf16_t sB[16384]; // 32 frags (8 ntile x 4 kc) * 512
  int tid = threadIdx.x;
  for (int i = tid; i < 16384; i += 256) {
    int frag = i >> 9, w = i & 511, ln = w >> 4, e = w & 15;
    int nt = frag >> 2, kc = frag & 3;
    int d = kc * 32 + kmap_e(ln, e);
    int n = nt * 16 + (ln & 15);
    sB[i] = (bf16_t)Bf[d * HID + n];
  }
  __syncthreads();

  int wave = tid >> 5, lane = tid & 31, m0 = lane & 15;
  int rowBase = blockIdx.x * 128 + wave * 16;
  int myRow = rowBase + m0; if (myRow >= M) myRow = M - 1;

  v8f acc[8] = {};
  for (int kc = 0; kc < 4; kc++) {
    Frag16 a; load_a_frag(a, A + (size_t)myRow * HID, lane, kc * 32);
    Frag16 b0, b1;
    load_b_frag(b0, sB, 0 * 4 + kc, lane);
    #pragma unroll
    for (int nt = 0; nt < 8; nt++) {
      Frag16& bc = (nt & 1) ? b1 : b0;
      if (nt < 7) {
        Frag16& bn = (nt & 1) ? b0 : b1;
        load_b_frag(bn, sB, (nt + 1) * 4 + kc, lane);
      }
      acc[nt] = wmma_bf16(a, bc, acc[nt]);
    }
  }
  bool hi = (lane & 16) != 0;
  for (int nt = 0; nt < 8; nt++)
    for (int r = 0; r < 8; r++) {
      int mrow = rowBase + (hi ? r + 8 : r);
      if (mrow < M) C[(size_t)mrow * HID + nt * 16 + m0] = (bf16_t)acc[nt][r];
    }
}

// ---------------------------------------------------------------------------
// NTN message kernel: per 16-edge tile, score = x_i @ Wb[s] . x_j fused via
// WMMA + 16-lane butterfly, alpha = tanh(score + [xi,et,xj]@Wl + bl),
// m = max(x_j, et)*alpha[slice] scatter-added into h[dst].
// Wb (all 4 slices) pre-swizzled into 128KB LDS; next tile's gathers are
// prefetched (global_prefetch_b8) while this tile's WMMAs run.
// ---------------------------------------------------------------------------
__global__ void __launch_bounds__(256, 1)
msg_kernel(const bf16_t* __restrict__ xt, const bf16_t* __restrict__ et,
           const int* __restrict__ eidx, const float* __restrict__ Wb,
           const float* __restrict__ Wl, const float* __restrict__ bl,
           float* __restrict__ hf)
{
  extern __shared__ char smem_raw[];
  bf16_t* sWb = (bf16_t*)smem_raw; // 65536 bf16 = 128 frags * 512
  int tid = threadIdx.x;
  for (int i = tid; i < 65536; i += 256) {
    int frag = i >> 9, w = i & 511, ln = w >> 4, e = w & 15;
    int s = frag >> 5, rem = frag & 31, ct = rem >> 2, kc = rem & 3;
    int d = kc * 32 + kmap_e(ln, e);
    int f = ct * 16 + (ln & 15);
    sWb[i] = (bf16_t)Wb[s * (HID * HID) + d * HID + f];
  }
  __syncthreads();

  int wave = tid >> 5, lane = tid & 31, m0 = lane & 15;
  bool hi = (lane & 16) != 0;
  float blv[SLICES];
  for (int s = 0; s < SLICES; s++) blv[s] = bl[s];

  for (int tile = blockIdx.x * 8 + wave; tile < EDGE_TILES; tile += gridDim.x * 8) {
    int Eb = tile * 16;

    // prefetch next tile's index block, et rows, and gathered xt rows
    int nextTile = tile + gridDim.x * 8;
    if (nextTile < EDGE_TILES) {
      int nEb = nextTile * 16;
      __builtin_prefetch(&eidx[nEb], 0, 1);
      __builtin_prefetch(&eidx[N_EDGES + nEb], 0, 1);
      __builtin_prefetch(&et[(size_t)nEb * HID + (size_t)lane * 64], 0, 1);
      int nSrc = eidx[nEb + m0];
      int nDst = eidx[N_EDGES + nEb + m0];
      __builtin_prefetch(&xt[(size_t)nSrc * HID], 0, 1);
      __builtin_prefetch(&xt[(size_t)nDst * HID], 0, 1);
    }

    // x_i = xt[dst] as WMMA A operand (row per lane = edge m0)
    int myDst = eidx[N_EDGES + Eb + m0];
    Frag16 ai[4];
    const bf16_t* rowi = xt + (size_t)myDst * HID;
    for (int kc = 0; kc < 4; kc++) load_a_frag(ai[kc], rowi, lane, kc * 32);

    // C-layout edge rows for this lane's half
    int src8[8], dst8[8];
    for (int r = 0; r < 8; r++) {
      int m = hi ? (r + 8) : r;
      src8[r] = eidx[Eb + m];
      dst8[r] = eidx[N_EDGES + Eb + m];
    }

    // x_j and et in C layout (lane holds column n across 8 edges), f32
    v8f xjc[8], etc8[8];
    for (int c2 = 0; c2 < 8; c2++) {
      int n = c2 * 16 + m0;
      for (int r = 0; r < 8; r++) {
        int m = hi ? (r + 8) : r;
        xjc[c2][r]  = (float)xt[(size_t)src8[r] * HID + n];
        etc8[c2][r] = (float)et[(size_t)(Eb + m) * HID + n];
      }
    }

    // bilinear score partials: T = x_i @ Wb[s], dotted with x_j per edge.
    // All 4 k-chunk B fragments preloaded (one clause) before the WMMA chain.
    v8f sc[4] = {};
    for (int s = 0; s < SLICES; s++) {
      for (int ct = 0; ct < 8; ct++) {
        Frag16 b[4];
        #pragma unroll
        for (int kc = 0; kc < 4; kc++)
          load_b_frag(b[kc], sWb, (s * 8 + ct) * 4 + kc, lane);
        v8f acc = {};
        #pragma unroll
        for (int kc = 0; kc < 4; kc++) acc = wmma_bf16(ai[kc], b[kc], acc);
        for (int r = 0; r < 8; r++) sc[s][r] += acc[r] * xjc[ct][r];
      }
    }

    // linear term: [x_i, et, x_j] @ Wl (per-column partials, same reduction)
    for (int c2 = 0; c2 < 8; c2++) {
      int n = c2 * 16 + m0;
      float w0[4], w1[4], w2v[4];
      for (int s = 0; s < 4; s++) {
        w0[s]  = Wl[n * 4 + s];
        w1[s]  = Wl[(HID + n) * 4 + s];
        w2v[s] = Wl[(2 * HID + n) * 4 + s];
      }
      for (int r = 0; r < 8; r++) {
        float xiv = (float)xt[(size_t)dst8[r] * HID + n];
        float ev = etc8[c2][r], jv = xjc[c2][r];
        for (int s = 0; s < 4; s++)
          sc[s][r] += xiv * w0[s] + ev * w1[s] + jv * w2v[s];
      }
    }

    // reduce over the 16 lanes of each half (columns), then alpha
    float alpha[4][8];
    for (int s = 0; s < 4; s++)
      for (int r = 0; r < 8; r++) {
        float v = sc[s][r];
        v += __shfl_xor(v, 1, 32);
        v += __shfl_xor(v, 2, 32);
        v += __shfl_xor(v, 4, 32);
        v += __shfl_xor(v, 8, 32);
        alpha[s][r] = tanhf(v + blv[s]);
      }

    // messages: m = max(x_j, et) * alpha[slice(n)], scatter by dst
    for (int c2 = 0; c2 < 8; c2++) {
      int s = c2 >> 1;            // slice = n/32, uniform per 16-col tile
      int n = c2 * 16 + m0;
      for (int r = 0; r < 8; r++) {
        float v = fmaxf(xjc[c2][r], etc8[c2][r]) * alpha[s][r];
        atomicAdd(&hf[(size_t)dst8[r] * HID + n], v);
      }
    }
  }
}

// ---------------------------------------------------------------------------
// Gate: beta = sigmoid(x@(G0+G2) + h@(G1-G2) + b); x = beta*x + (1-beta)*h.
// Folded weights built during LDS staging; B-fragments double-buffered.
// Updates xf and xb in place.
// ---------------------------------------------------------------------------
__global__ void __launch_bounds__(256, 1)
gate_kernel(bf16_t* __restrict__ xb, const bf16_t* __restrict__ hb,
            const float* __restrict__ gw, const float* __restrict__ gb,
            float* __restrict__ xf, const float* __restrict__ hf, int M)
{
  extern __shared__ char smem_raw[];
  bf16_t* sW = (bf16_t*)smem_raw; // [2][16384] swizzled frags (gA, gB)
  int tid = threadIdx.x;
  for (int i = tid; i < 32768; i += 256) {
    int buf = i >> 14, j = i & 16383;
    int frag = j >> 9, w = j & 511, ln = w >> 4, e = w & 15;
    int nt = frag >> 2, kc = frag & 3;
    int d = kc * 32 + kmap_e(ln, e);
    int n = nt * 16 + (ln & 15);
    float v = (buf == 0) ? (gw[d * HID + n] + gw[(2 * HID + d) * HID + n])
                         : (gw[(HID + d) * HID + n] - gw[(2 * HID + d) * HID + n]);
    sW[i] = (bf16_t)v;
  }
  __syncthreads();

  int wave = tid >> 5, lane = tid & 31, m0 = lane & 15;
  int rowBase = blockIdx.x * 128 + wave * 16;
  int myRow = rowBase + m0; if (myRow >= M) myRow = M - 1;

  v8f acc[8] = {};
  for (int kc = 0; kc < 4; kc++) {
    Frag16 ax, ah;
    load_a_frag(ax, xb + (size_t)myRow * HID, lane, kc * 32);
    load_a_frag(ah, hb + (size_t)myRow * HID, lane, kc * 32);
    Frag16 bA0, bB0, bA1, bB1;
    load_b_frag(bA0, sW, 0 * 4 + kc, lane);
    load_b_frag(bB0, sW + 16384, 0 * 4 + kc, lane);
    #pragma unroll
    for (int nt = 0; nt < 8; nt++) {
      Frag16& bAc = (nt & 1) ? bA1 : bA0;
      Frag16& bBc = (nt & 1) ? bB1 : bB0;
      if (nt < 7) {
        Frag16& bAn = (nt & 1) ? bA0 : bA1;
        Frag16& bBn = (nt & 1) ? bB0 : bB1;
        load_b_frag(bAn, sW, (nt + 1) * 4 + kc, lane);
        load_b_frag(bBn, sW + 16384, (nt + 1) * 4 + kc, lane);
      }
      acc[nt] = wmma_bf16(ax, bAc, acc[nt]);
      acc[nt] = wmma_bf16(ah, bBc, acc[nt]);
    }
  }
  bool hi = (lane & 16) != 0;
  for (int nt = 0; nt < 8; nt++) {
    int n = nt * 16 + m0;
    float gbn = gb[n];
    for (int r = 0; r < 8; r++) {
      int mrow = rowBase + (hi ? r + 8 : r);
      if (mrow < M) {
        float beta = 1.0f / (1.0f + expf(-(acc[nt][r] + gbn)));
        size_t idx = (size_t)mrow * HID + n;
        float xo = xf[idx], ho = hf[idx];
        float xn = beta * xo + (1.0f - beta) * ho;
        xf[idx] = xn;
        xb[idx] = (bf16_t)xn;
      }
    }
  }
}

// ---------------------------------------------------------------------------
// Small / elementwise kernels
// ---------------------------------------------------------------------------
__global__ void lin_node_kernel(const float* __restrict__ x, const float* __restrict__ W,
                                const float* __restrict__ b, float* __restrict__ xf,
                                bf16_t* __restrict__ xb)
{
  __shared__ float sx[IN_CH];
  int nid = blockIdx.x, t = threadIdx.x;
  if (t < IN_CH) sx[t] = x[(size_t)nid * IN_CH + t];
  __syncthreads();
  float acc = b[t];
  for (int k = 0; k < IN_CH; k++) acc += sx[k] * W[k * HID + t];
  acc = fmaxf(acc, 0.0f);
  size_t idx = (size_t)nid * HID + t;
  xf[idx] = acc; xb[idx] = (bf16_t)acc;
}

__global__ void lin_edge_kernel(const float* __restrict__ ea, const float* __restrict__ W,
                                const float* __restrict__ b, bf16_t* __restrict__ eb)
{
  __shared__ float se[EDGE_DIM];
  int eid = blockIdx.x, t = threadIdx.x;
  if (t < EDGE_DIM) se[t] = ea[(size_t)eid * EDGE_DIM + t];
  __syncthreads();
  float acc = b[t];
  for (int k = 0; k < EDGE_DIM; k++) acc += se[k] * W[k * HID + t];
  eb[(size_t)eid * HID + t] = (bf16_t)fmaxf(acc, 0.0f);
}

__global__ void fill_kernel(float* p, float v, int n) {
  int i = blockIdx.x * blockDim.x + threadIdx.x;
  if (i < n) p[i] = v;
}

__global__ void relu_store_kernel(float* __restrict__ hf, bf16_t* __restrict__ hb, int n) {
  int i = blockIdx.x * blockDim.x + threadIdx.x;
  if (i >= n) return;
  float v = fmaxf(hf[i], 0.0f);
  hf[i] = v; hb[i] = (bf16_t)v;
}

__global__ void fa_scatter_kernel(const float* __restrict__ xf, const int* __restrict__ batch,
                                  float* __restrict__ gmax, float* __restrict__ gsum)
{
  int nid = blockIdx.x, c = threadIdx.x;
  int g = batch[nid];
  float v = xf[(size_t)nid * HID + c];
  atomicAdd(&gsum[g * HID + c], v);
  atomicMaxFloat(&gmax[g * HID + c], v);
}

__global__ void fa_mlp_kernel(const float* __restrict__ gmax, const float* __restrict__ gsum,
                              const float* __restrict__ w1, const float* __restrict__ w2,
                              float* __restrict__ y)
{
  __shared__ float smx[HID], ssm[HID], hm[FA_HID], hs[FA_HID];
  int g = blockIdx.x, c = threadIdx.x;
  float mv = gmax[g * HID + c];
  if (!__builtin_isfinite(mv)) mv = 0.0f;   // empty segments -> 0
  smx[c] = mv;
  ssm[c] = gsum[g * HID + c];
  __syncthreads();
  if (c < FA_HID) {
    float a = 0.f, b = 0.f;
    for (int k = 0; k < HID; k++) { a += smx[k] * w1[k * FA_HID + c]; b += ssm[k] * w1[k * FA_HID + c]; }
    hm[c] = fmaxf(a, 0.f); hs[c] = fmaxf(b, 0.f);
  }
  __syncthreads();
  float acc = 0.f;
  for (int k = 0; k < FA_HID; k++) acc += (hm[k] + hs[k]) * w2[k * HID + c];
  y[g * HID + c] = 1.0f / (1.0f + expf(-acc));
}

__global__ void fa_scale_kernel(float* __restrict__ xf, bf16_t* __restrict__ xb,
                                const int* __restrict__ batch, const float* __restrict__ y, int n)
{
  int i = blockIdx.x * blockDim.x + threadIdx.x;
  if (i >= n) return;
  int node = i >> 7, c = i & 127;
  float v = xf[i] * y[batch[node] * HID + c];
  xf[i] = v; xb[i] = (bf16_t)v;
}

__global__ void mol_scatter_kernel(const float* __restrict__ xf, const int* __restrict__ batch,
                                   float* __restrict__ mol)
{
  int nid = blockIdx.x, c = threadIdx.x;
  atomicAdd(&mol[batch[nid] * HID + c], xf[(size_t)nid * HID + c]);
}

__global__ void out_kernel(const float* __restrict__ mol, const float* __restrict__ ow,
                           const float* __restrict__ ob, float* __restrict__ out)
{
  __shared__ float red[HID];
  int g = blockIdx.x, c = threadIdx.x;
  red[c] = fmaxf(mol[g * HID + c], 0.0f) * ow[c];
  __syncthreads();
  for (int s = HID / 2; s > 0; s >>= 1) {
    if (c < s) red[c] += red[c + s];
    __syncthreads();
  }
  if (c == 0) out[g] = red[0] + ob[0];
}

// ---------------------------------------------------------------------------
extern "C" void kernel_launch(void* const* d_in, const int* in_sizes, int n_in,
                              void* d_out, int out_size, void* d_ws, size_t ws_size,
                              hipStream_t stream)
{
  (void)in_sizes; (void)n_in; (void)out_size; (void)ws_size;
  const float* x_in    = (const float*)d_in[0];
  const int*   eidx    = (const int*)  d_in[1];
  const float* eattr   = (const float*)d_in[2];
  const int*   batch   = (const int*)  d_in[3];
  const float* lin_a_w = (const float*)d_in[4];
  const float* lin_a_b = (const float*)d_in[5];
  const float* lin_b_w = (const float*)d_in[6];
  const float* lin_b_b = (const float*)d_in[7];
  const float* Wn      = (const float*)d_in[8];
  const float* We      = (const float*)d_in[9];
  const float* Wb      = (const float*)d_in[10];
  const float* Wl      = (const float*)d_in[11];
  const float* bl      = (const float*)d_in[12];
  const float* gate_w  = (const float*)d_in[13];
  const float* gate_b  = (const float*)d_in[14];
  const float* fa_w1   = (const float*)d_in[15];
  const float* fa_w2   = (const float*)d_in[16];
  const float* out_w   = (const float*)d_in[17];
  const float* out_b   = (const float*)d_in[18];
  float* out = (float*)d_out;

  char* ws = (char*)d_ws;
  size_t off = 0;
  auto take = [&](size_t bytes) -> char* {
    char* p = ws + off;
    off = (off + bytes + 255) & ~(size_t)255;
    return p;
  };
  float*  xf   = (float*) take((size_t)N_NODES * HID * 4);
  bf16_t* xb   = (bf16_t*)take((size_t)N_NODES * HID * 2);
  float*  hf   = (float*) take((size_t)N_NODES * HID * 4);
  bf16_t* hb   = (bf16_t*)take((size_t)N_NODES * HID * 2);
  bf16_t* xtb  = (bf16_t*)take((size_t)N_NODES * HID * 2);
  bf16_t* eab  = (bf16_t*)take((size_t)N_EDGES * HID * 2);
  bf16_t* etb  = (bf16_t*)take((size_t)N_EDGES * HID * 2);
  float*  gmax = (float*) take((size_t)N_GRAPHS * HID * 4);
  float*  gsum = (float*) take((size_t)N_GRAPHS * HID * 4);
  float*  yatt = (float*) take((size_t)N_GRAPHS * HID * 4);
  float*  mol  = (float*) take((size_t)N_GRAPHS * HID * 4);

  lin_node_kernel<<<N_NODES, HID, 0, stream>>>(x_in, lin_a_w, lin_a_b, xf, xb);
  lin_edge_kernel<<<N_EDGES, HID, 0, stream>>>(eattr, lin_b_w, lin_b_b, eab);

  const int nodeBlocks = (N_NODES + 127) / 128;
  const int edgeBlocks = (N_EDGES + 127) / 128;
  const int elems = N_NODES * HID;

  for (int L = 0; L < 3; L++) {
    gemm128_kernel<<<nodeBlocks, 256, 0, stream>>>(xb, Wn + (size_t)L * HID * HID, xtb, N_NODES);
    gemm128_kernel<<<edgeBlocks, 256, 0, stream>>>(eab, We + (size_t)L * HID * HID, etb, N_EDGES);
    hipMemsetAsync(hf, 0, (size_t)N_NODES * HID * 4, stream);
    msg_kernel<<<512, 256, 131072, stream>>>(xtb, etb, eidx,
        Wb + (size_t)L * SLICES * HID * HID,
        Wl + (size_t)L * 3 * HID * SLICES,
        bl + (size_t)L * SLICES, hf);
    relu_store_kernel<<<(elems + 255) / 256, 256, 0, stream>>>(hf, hb, elems);
    gate_kernel<<<nodeBlocks, 256, 65536, stream>>>(xb, hb, gate_w, gate_b, xf, hf, N_NODES);
    fill_kernel<<<(N_GRAPHS * HID + 255) / 256, 256, 0, stream>>>(gmax, -__builtin_huge_valf(), N_GRAPHS * HID);
    hipMemsetAsync(gsum, 0, (size_t)N_GRAPHS * HID * 4, stream);
    fa_scatter_kernel<<<N_NODES, HID, 0, stream>>>(xf, batch, gmax, gsum);
    fa_mlp_kernel<<<N_GRAPHS, HID, 0, stream>>>(gmax, gsum, fa_w1, fa_w2, yatt);
    fa_scale_kernel<<<(elems + 255) / 256, 256, 0, stream>>>(xf, xb, batch, yatt, elems);
  }

  hipMemsetAsync(mol, 0, (size_t)N_GRAPHS * HID * 4, stream);
  mol_scatter_kernel<<<N_NODES, HID, 0, stream>>>(xf, batch, mol);
  out_kernel<<<N_GRAPHS, HID, 0, stream>>>(mol, out_w, out_b, out);
}
